// MinVQVAE_12902081757256
// MI455X (gfx1250) — matrix operand order repaired and checked by reading, not compiled, not run on hardware
//
#include <hip/hip_runtime.h>
#include <hip/hip_bf16.h>
#include <math.h>

// ---------------------------------------------------------------------------
// MinVQVAE forward on gfx1250 (MI455X): WMMA f16 GEMM pipeline, f32 accumulate
//  - all weights pre-transposed/padded so LDS fragment reads are ds_load_b128
//  - double-buffered LDS staging; async global->LDS (ASYNCcnt) data movement
// ---------------------------------------------------------------------------

typedef _Float16 half_t;
typedef __attribute__((ext_vector_type(16))) _Float16 v16h;
typedef __attribute__((ext_vector_type(8)))  _Float16 v8h;
typedef __attribute__((ext_vector_type(8)))  float    v8f;
typedef __attribute__((ext_vector_type(4)))  int      v4i;

#define NN 32768   // batch
#define DD 784     // input dim
#define DP 800     // input dim padded to multiple of 32 (K dim of GEMM1)
#define KK 4096    // n_category
#define LL 64      // latent dim
#define HH 512     // hidden dim

// gfx1250 async global->LDS (ASYNCcnt) path, graceful fallback if absent
#if defined(__HIP_DEVICE_COMPILE__) && \
    __has_builtin(__builtin_amdgcn_global_load_async_to_lds_b128) && \
    __has_builtin(__builtin_amdgcn_s_wait_asynccnt)
#define HAVE_ASYNC_LDS 1
#else
#define HAVE_ASYNC_LDS 0
#endif

#if HAVE_ASYNC_LDS
typedef __attribute__((address_space(1))) v4i* gv4i_p;
typedef __attribute__((address_space(3))) v4i* lv4i_p;
#endif

__device__ __forceinline__ void cp16(half_t* lds_dst, const half_t* g_src) {
#if HAVE_ASYNC_LDS
    __builtin_amdgcn_global_load_async_to_lds_b128(
        (gv4i_p)g_src, (lv4i_p)lds_dst, 0, 0);
#else
    *(v8h*)lds_dst = *(const v8h*)g_src;
#endif
}
__device__ __forceinline__ void cp_fence() {
#if HAVE_ASYNC_LDS
    __builtin_amdgcn_s_wait_asynccnt(0);
#endif
}

__device__ __forceinline__ v16h mk16(v8h lo, v8h hi8) {
    v16h r;
    #pragma unroll
    for (int j = 0; j < 8; ++j) { r[j] = lo[j]; r[j + 8] = hi8[j]; }
    return r;
}

__device__ __forceinline__ float gelu_exact(float x) {
    return 0.5f * x * (1.0f + erff(x * 0.7071067811865476f));
}
__device__ __forceinline__ float sigmoidf_(float x) {
    return 1.0f / (1.0f + expf(-x));
}

// ---------------------------------------------------------------------------
// prep kernels (bandwidth-trivial)
// ---------------------------------------------------------------------------

// x: [N,784] f32 -> [N,800] f16, zero padded columns
__global__ void cvt_pad_cols(const float* __restrict__ in, half_t* __restrict__ out,
                             int rows, int cols, int cols_pad) {
    size_t n = (size_t)rows * cols_pad;
    for (size_t i = (size_t)blockIdx.x * blockDim.x + threadIdx.x; i < n;
         i += (size_t)gridDim.x * blockDim.x) {
        int r = (int)(i / cols_pad);
        int c = (int)(i % cols_pad);
        out[i] = (c < cols) ? (half_t)in[(size_t)r * cols + c] : (half_t)0.0f;
    }
}

// W: [K,N] f32 -> WT: [Npad,Kpad] f16 (transposed, zero padded both dims)
__global__ void cvt_transpose_pad(const float* __restrict__ in, half_t* __restrict__ out,
                                  int K, int N, int Kpad, int Npad) {
    size_t n = (size_t)Npad * Kpad;
    for (size_t i = (size_t)blockIdx.x * blockDim.x + threadIdx.x; i < n;
         i += (size_t)gridDim.x * blockDim.x) {
        int nn = (int)(i / Kpad);
        int k  = (int)(i % Kpad);
        out[i] = (nn < N && k < K) ? (half_t)in[(size_t)k * N + nn] : (half_t)0.0f;
    }
}

// embed_pool [K,L] f32 -> f16 row-major copy + per-row squared norms
__global__ void embed_prep(const float* __restrict__ ep, half_t* __restrict__ eph,
                           float* __restrict__ epn) {
    int k = blockIdx.x * blockDim.x + threadIdx.x;
    if (k >= KK) return;
    float s = 0.0f;
    #pragma unroll 4
    for (int l = 0; l < LL; ++l) {
        float v = ep[(size_t)k * LL + l];
        s += v * v;
        eph[(size_t)k * LL + l] = (half_t)v;
    }
    epn[k] = s;
}

__global__ void gather_zq(const float* __restrict__ ep, const int* __restrict__ idx,
                          half_t* __restrict__ zq) {
    size_t n = (size_t)NN * LL;
    for (size_t i = (size_t)blockIdx.x * blockDim.x + threadIdx.x; i < n;
         i += (size_t)gridDim.x * blockDim.x) {
        int r = (int)(i >> 6);
        int l = (int)(i & 63);
        zq[i] = (half_t)ep[(size_t)idx[r] * LL + l];
    }
}

__global__ void onehot_fill(const int* __restrict__ idx, float* __restrict__ out) {
    size_t n = (size_t)NN * KK;
    for (size_t i = (size_t)blockIdx.x * blockDim.x + threadIdx.x; i < n;
         i += (size_t)gridDim.x * blockDim.x) {
        int r = (int)(i >> 12);           // KK = 4096
        int c = (int)(i & (KK - 1));
        out[i] = (c == idx[r]) ? 1.0f : 0.0f;
    }
}

__global__ void loss_init(float* __restrict__ out) {
    if (threadIdx.x == 0 && blockIdx.x == 0) *out = 0.0f;
}

__global__ void sqdiff_reduce(const float* __restrict__ a, const float* __restrict__ b,
                              float scale, float* __restrict__ out, size_t n) {
    __shared__ float sd[256];
    float s = 0.0f;
    for (size_t i = (size_t)blockIdx.x * blockDim.x + threadIdx.x; i < n;
         i += (size_t)gridDim.x * blockDim.x) {
        float d = a[i] - b[i];
        s += d * d;
    }
    sd[threadIdx.x] = s;
    __syncthreads();
    for (int st = 128; st > 0; st >>= 1) {
        if ((int)threadIdx.x < st) sd[threadIdx.x] += sd[threadIdx.x + st];
        __syncthreads();
    }
    if (threadIdx.x == 0) atomicAdd(out, sd[0] * scale);
}

__global__ void vq_loss_reduce(const float* __restrict__ ze, const float* __restrict__ ep,
                               const int* __restrict__ idx, float scale,
                               float* __restrict__ out) {
    __shared__ float sd[256];
    size_t n = (size_t)NN * LL;
    float s = 0.0f;
    for (size_t i = (size_t)blockIdx.x * blockDim.x + threadIdx.x; i < n;
         i += (size_t)gridDim.x * blockDim.x) {
        int r = (int)(i >> 6);
        int l = (int)(i & 63);
        float d = ze[i] - ep[(size_t)idx[r] * LL + l];
        s += d * d;
    }
    sd[threadIdx.x] = s;
    __syncthreads();
    for (int st = 128; st > 0; st >>= 1) {
        if ((int)threadIdx.x < st) sd[threadIdx.x] += sd[threadIdx.x + st];
        __syncthreads();
    }
    if (threadIdx.x == 0) atomicAdd(out, sd[0] * scale);
}

// ---------------------------------------------------------------------------
// WMMA GEMM:  C[M,N] = act(A[M,K] * B + bias), B given pre-transposed:
//   BT [Npad,K] f16 row-major, Npad multiple of 128, zero-padded.
//   Block = 256 threads (8 waves), 64x128 tile; wave (wm,wn) owns 32x32,
//   i.e. 4 x v_wmma_f32_16x16x32_f16 per 32-wide K step. Double-buffered LDS.
//   EPI: 0 = GELU -> f16 | 1 = identity -> f16 + f32 | 2 = sigmoid -> f32
// ---------------------------------------------------------------------------
template <int EPI>
__global__ __launch_bounds__(256)
void gemm_bias_act(const half_t* __restrict__ A, const half_t* __restrict__ BT,
                   const float* __restrict__ bias,
                   half_t* __restrict__ Ch, float* __restrict__ Cf,
                   int M, int K, int N) {
    __shared__ half_t sA[2][64][40];    // 32 k + 8 pad halves per row
    __shared__ half_t sB[2][128][40];

    const int tid  = threadIdx.x;
    const int wave = tid >> 5;
    const int lane = tid & 31;
    const int hi   = lane >> 4;
    const int ln16 = lane & 15;
    const int wm   = wave & 1;          // 0..1 -> 32-row half
    const int wn   = wave >> 1;         // 0..3 -> 32-col quarter
    const int bm   = blockIdx.x * 64;
    const int bn   = blockIdx.y * 128;

    const int ar = tid >> 2;            // 0..63
    const int cg = tid & 3;             // 0..3 (8 halves each)

    v8f acc[2][2] = {};

    // prologue: stage k-tile 0 into buffer 0
    cp16(&sA[0][ar][cg * 8], A + (size_t)(bm + ar) * K + cg * 8);
    cp16(&sB[0][ar][cg * 8], BT + (size_t)(bn + ar) * K + cg * 8);
    cp16(&sB[0][64 + ar][cg * 8], BT + (size_t)(bn + 64 + ar) * K + cg * 8);

    const int nk = K >> 5;
    for (int ki = 0; ki < nk; ++ki) {
        const int buf = ki & 1;
        cp_fence();
        __syncthreads();
        if (ki + 1 < nk) {
            const int k1 = (ki + 1) * 32;
            cp16(&sA[buf ^ 1][ar][cg * 8], A + (size_t)(bm + ar) * K + k1 + cg * 8);
            cp16(&sB[buf ^ 1][ar][cg * 8], BT + (size_t)(bn + ar) * K + k1 + cg * 8);
            cp16(&sB[buf ^ 1][64 + ar][cg * 8], BT + (size_t)(bn + 64 + ar) * K + k1 + cg * 8);
        }
        // fragments: contiguous 8-half runs -> ds_load_b128 pairs
        v16h a[2], b[2];
        #pragma unroll
        for (int mi = 0; mi < 2; ++mi) {
            const int mrow = wm * 32 + mi * 16 + ln16;
            a[mi] = mk16(*(const v8h*)&sA[buf][mrow][hi * 8],
                         *(const v8h*)&sA[buf][mrow][16 + hi * 8]);
        }
        #pragma unroll
        for (int ni = 0; ni < 2; ++ni) {
            const int nrow = wn * 32 + ni * 16 + ln16;
            b[ni] = mk16(*(const v8h*)&sB[buf][nrow][hi * 8],
                         *(const v8h*)&sB[buf][nrow][16 + hi * 8]);
        }
        #pragma unroll
        for (int mi = 0; mi < 2; ++mi)
            #pragma unroll
            for (int ni = 0; ni < 2; ++ni)
                acc[mi][ni] = __builtin_amdgcn_wmma_f32_16x16x32_f16(
                    false, a[mi], false, b[ni], (short)0, acc[mi][ni], false, false);
    }

    // epilogue: D layout m = r + 8*hi, n = lane%16
    #pragma unroll
    for (int mi = 0; mi < 2; ++mi) {
        #pragma unroll
        for (int ni = 0; ni < 2; ++ni) {
            #pragma unroll
            for (int r = 0; r < 8; ++r) {
                const int gm = bm + wm * 32 + mi * 16 + r + hi * 8;
                const int gn = bn + wn * 32 + ni * 16 + ln16;
                if (gn < N) {
                    float v = acc[mi][ni][r] + bias[gn];
                    if constexpr (EPI == 0) {
                        Ch[(size_t)gm * N + gn] = (half_t)gelu_exact(v);
                    } else if constexpr (EPI == 1) {
                        Ch[(size_t)gm * N + gn] = (half_t)v;
                        Cf[(size_t)gm * N + gn] = v;
                    } else {
                        Cf[(size_t)gm * N + gn] = sigmoidf_(v);
                    }
                }
            }
        }
    }
}

// ---------------------------------------------------------------------------
// Codebook search: argmin_k ( |e_k|^2 - 2 z_e.e_k ) via WMMA.
//   B of this GEMM is E^T, so its pre-transposed form is E itself (row-major):
//   all LDS fragment reads are contiguous b128 pairs. Double-buffered chunks.
// ---------------------------------------------------------------------------
__global__ __launch_bounds__(256)
void vq_argmin(const half_t* __restrict__ ZE,   // [NN, 64] f16
               const half_t* __restrict__ EPH,  // [KK, 64] f16 (row-major E)
               const float* __restrict__ EPN,   // [KK] |e_k|^2
               int* __restrict__ ZIDX) {
    __shared__ half_t sZ[128][72];      // 128 rows x 64 latent (+8 pad)
    __shared__ half_t sE[2][64][72];    // 64-code chunk x 64 latent (+8 pad)

    const int tid  = threadIdx.x;
    const int wave = tid >> 5;
    const int lane = tid & 31;
    const int hi   = lane >> 4;
    const int ln16 = lane & 15;
    const int bm   = blockIdx.x * 128;

    // stage 128 z_e rows (16B copies)
    for (int i = tid; i < 128 * 8; i += 256) {
        const int r = i >> 3, g = i & 7;
        cp16(&sZ[r][g * 8], ZE + (size_t)(bm + r) * LL + g * 8);
    }
    // stage first codebook chunk
    for (int i = tid; i < 64 * 8; i += 256) {
        const int r = i >> 3, g = i & 7;
        cp16(&sE[0][r][g * 8], EPH + (size_t)r * LL + g * 8);
    }
    cp_fence();
    __syncthreads();

    // A fragments for this wave's 16 rows (k = 0..31 and 32..63)
    v16h a0, a1;
    {
        const int mrow = wave * 16 + ln16;
        a0 = mk16(*(const v8h*)&sZ[mrow][hi * 8],
                  *(const v8h*)&sZ[mrow][16 + hi * 8]);
        a1 = mk16(*(const v8h*)&sZ[mrow][32 + hi * 8],
                  *(const v8h*)&sZ[mrow][48 + hi * 8]);
    }

    float minv[8];
    int   mini[8];
    #pragma unroll
    for (int r = 0; r < 8; ++r) { minv[r] = 3.4e38f; mini[r] = 0; }

    const int nchunk = KK / 64;
    for (int it = 0; it < nchunk; ++it) {
        const int buf = it & 1;
        if (it + 1 < nchunk) {
            const int c1 = (it + 1) * 64;
            for (int i = tid; i < 64 * 8; i += 256) {
                const int r = i >> 3, g = i & 7;
                cp16(&sE[buf ^ 1][r][g * 8], EPH + (size_t)(c1 + r) * LL + g * 8);
            }
        }
        #pragma unroll
        for (int t = 0; t < 4; ++t) {
            const int nc = t * 16 + ln16;
            v16h b0 = mk16(*(const v8h*)&sE[buf][nc][hi * 8],
                           *(const v8h*)&sE[buf][nc][16 + hi * 8]);
            v16h b1 = mk16(*(const v8h*)&sE[buf][nc][32 + hi * 8],
                           *(const v8h*)&sE[buf][nc][48 + hi * 8]);
            v8f acc = {};
            acc = __builtin_amdgcn_wmma_f32_16x16x32_f16(false, a0, false, b0,
                                                         (short)0, acc, false, false);
            acc = __builtin_amdgcn_wmma_f32_16x16x32_f16(false, a1, false, b1,
                                                         (short)0, acc, false, false);
            const int gn = it * 64 + t * 16 + ln16;
            const float en = EPN[gn];
            #pragma unroll
            for (int r = 0; r < 8; ++r) {
                float d = en - 2.0f * acc[r];
                if (d < minv[r]) { minv[r] = d; mini[r] = gn; }
            }
        }
        cp_fence();
        __syncthreads();
    }

    // reduce across the 16 lanes of each half (wave32: xor masks 1,2,4,8)
    #pragma unroll
    for (int r = 0; r < 8; ++r) {
        float v = minv[r];
        int   ix = mini[r];
        #pragma unroll
        for (int m = 8; m >= 1; m >>= 1) {
            float ov = __shfl_xor(v, m, 32);
            int   oi = __shfl_xor(ix, m, 32);
            if (ov < v || (ov == v && oi < ix)) { v = ov; ix = oi; }
        }
        if (ln16 == 0) ZIDX[bm + wave * 16 + r + hi * 8] = ix;
    }
}

// ---------------------------------------------------------------------------
// launch
// ---------------------------------------------------------------------------
extern "C" void kernel_launch(void* const* d_in, const int* in_sizes, int n_in,
                              void* d_out, int out_size, void* d_ws, size_t ws_size,
                              hipStream_t stream) {
    (void)in_sizes; (void)n_in; (void)out_size; (void)ws_size;

    const float* x      = (const float*)d_in[0];
    const float* ep     = (const float*)d_in[1];
    const float* enc_w1 = (const float*)d_in[2];
    const float* enc_b1 = (const float*)d_in[3];
    const float* enc_w2 = (const float*)d_in[4];
    const float* enc_b2 = (const float*)d_in[5];
    const float* enc_w3 = (const float*)d_in[6];
    const float* enc_b3 = (const float*)d_in[7];
    const float* dec_w1 = (const float*)d_in[8];
    const float* dec_b1 = (const float*)d_in[9];
    const float* dec_w2 = (const float*)d_in[10];
    const float* dec_b2 = (const float*)d_in[11];
    const float* dec_w3 = (const float*)d_in[12];
    const float* dec_b3 = (const float*)d_in[13];

    // workspace carve-out (256B aligned)
    char* ws = (char*)d_ws;
    size_t off = 0;
    auto carve = [&](size_t bytes) {
        void* p = ws + off;
        off = (off + bytes + 255) & ~(size_t)255;
        return p;
    };
    // BT buffers: [Npad][Kpad], Npad multiple of 128, zero padded
    half_t* XH   = (half_t*)carve((size_t)NN * DP * 2);      // [N,800]
    half_t* W1T  = (half_t*)carve((size_t)512 * DP * 2);     // [512][800]
    half_t* W2T  = (half_t*)carve((size_t)512 * HH * 2);     // [512][512]
    half_t* W3T  = (half_t*)carve((size_t)128 * HH * 2);     // [128][512]
    half_t* DW1T = (half_t*)carve((size_t)512 * LL * 2);     // [512][64]
    half_t* DW2T = (half_t*)carve((size_t)512 * HH * 2);     // [512][512]
    half_t* DW3T = (half_t*)carve((size_t)896 * HH * 2);     // [896][512]
    half_t* EPH  = (half_t*)carve((size_t)KK * LL * 2);      // [4096][64]
    float*  EPN  = (float*)carve((size_t)KK * 4);
    half_t* H1   = (half_t*)carve((size_t)NN * HH * 2);
    half_t* H2   = (half_t*)carve((size_t)NN * HH * 2);
    float*  ZEF  = (float*)carve((size_t)NN * LL * 4);
    half_t* ZEH  = (half_t*)carve((size_t)NN * LL * 2);
    half_t* ZQH  = (half_t*)carve((size_t)NN * LL * 2);
    int*    ZIDX = (int*)carve((size_t)NN * 4);

    float* x_pred = (float*)d_out;
    float* oneh   = (float*)d_out + (size_t)NN * DD;
    float* loss   = (float*)d_out + (size_t)NN * DD + (size_t)NN * KK;

    const int T = 256;
    auto gs = [&](size_t n) {
        size_t b = (n + T - 1) / T;
        return (unsigned)(b > 262144 ? 262144 : b);
    };

    // 1) precision conversion: pad x, transpose+pad all weights, prep codebook
    cvt_pad_cols<<<gs((size_t)NN * DP), T, 0, stream>>>(x, XH, NN, DD, DP);
    cvt_transpose_pad<<<gs((size_t)512 * DP), T, 0, stream>>>(enc_w1, W1T, DD, HH, DP, 512);
    cvt_transpose_pad<<<gs((size_t)512 * HH), T, 0, stream>>>(enc_w2, W2T, HH, HH, HH, 512);
    cvt_transpose_pad<<<gs((size_t)128 * HH), T, 0, stream>>>(enc_w3, W3T, HH, LL, HH, 128);
    cvt_transpose_pad<<<gs((size_t)512 * LL), T, 0, stream>>>(dec_w1, DW1T, LL, HH, LL, 512);
    cvt_transpose_pad<<<gs((size_t)512 * HH), T, 0, stream>>>(dec_w2, DW2T, HH, HH, HH, 512);
    cvt_transpose_pad<<<gs((size_t)896 * HH), T, 0, stream>>>(dec_w3, DW3T, HH, DD, HH, 896);
    embed_prep<<<KK / T, T, 0, stream>>>(ep, EPH, EPN);

    // 2) encoder
    gemm_bias_act<0><<<dim3(NN / 64, 4), T, 0, stream>>>(XH, W1T, enc_b1, H1, nullptr, NN, DP, HH);
    gemm_bias_act<0><<<dim3(NN / 64, 4), T, 0, stream>>>(H1, W2T, enc_b2, H2, nullptr, NN, HH, HH);
    gemm_bias_act<1><<<dim3(NN / 64, 1), T, 0, stream>>>(H2, W3T, enc_b3, ZEH, ZEF, NN, HH, LL);

    // 3) codebook search + quantize
    vq_argmin<<<NN / 128, T, 0, stream>>>(ZEH, EPH, EPN, ZIDX);
    gather_zq<<<gs((size_t)NN * LL), T, 0, stream>>>(ep, ZIDX, ZQH);
    onehot_fill<<<gs((size_t)NN * KK), T, 0, stream>>>(ZIDX, oneh);

    // 4) decoder (straight-through: forward uses z_q)
    gemm_bias_act<0><<<dim3(NN / 64, 4), T, 0, stream>>>(ZQH, DW1T, dec_b1, H1, nullptr, NN, LL, HH);
    gemm_bias_act<0><<<dim3(NN / 64, 4), T, 0, stream>>>(H1, DW2T, dec_b2, H2, nullptr, NN, HH, HH);
    gemm_bias_act<2><<<dim3(NN / 64, 7), T, 0, stream>>>(H2, DW3T, dec_b3, nullptr, x_pred, NN, HH, DD);

    // 5) loss = [mean((x-x_pred)^2) + 1.25*mean((z_e-z_q)^2)] / N
    loss_init<<<1, 1, 0, stream>>>(loss);
    float recon_scale = 1.0f / ((float)NN * (float)DD * (float)NN);
    float vq_scale    = 1.25f / ((float)NN * (float)LL * (float)NN);
    sqdiff_reduce<<<2048, T, 0, stream>>>(x, x_pred, recon_scale, loss, (size_t)NN * DD);
    vq_loss_reduce<<<2048, T, 0, stream>>>(ZEF, ep, ZIDX, vq_scale, loss);
}
